// _GRUD_6846177870512
// MI455X (gfx1250) — matrix operand order repaired and checked by reading, not compiled, not run on hardware
//
#include <hip/hip_runtime.h>
#include <hip/hip_bf16.h>
#include <math.h>

// GRU-D forward for MI455X (gfx1250, wave32, WMMA bf16 16x16x32, f32 accum).
//
// Shapes: B=2048, T=200, F=64, H=128, C=1, IN=2F+H=256.
// Grid: 128 workgroups x 256 threads (8 waves). Each WG owns 16 batch rows,
// keeps h in LDS across all 200 timesteps. Wave w owns H columns [16w,16w+16).
// Weights are pre-packed to bf16 WMMA B-fragments; the compiler keeps them
// register-resident across the time loop (verified in round-1 asm).
// Per-step input staging uses double-buffered GLOBAL_LOAD_ASYNC_TO_LDS_B128
// (ASYNCcnt) when the toolchain exposes the builtins.

typedef __attribute__((ext_vector_type(16))) __bf16 v16bf;
typedef __attribute__((ext_vector_type(8)))  float v8f;

static constexpr int Bsz = 2048;
static constexpr int Tt  = 200;
static constexpr int Ff  = 64;
static constexpr int Hh  = 128;
static constexpr int INs = 256;   // 2F + H

// Packed-weight tile counts (each tile = 32 lanes * 16 halfs = 512 bf16 = 1KB)
static constexpr int TILES_IH = 24 * 8;   // w_ih [384,256]: 24 n-tiles x 8 k-blocks
static constexpr int TILES_HH = 24 * 4;   // w_hh [384,128]: 24 n-tiles x 4 k-blocks
static constexpr int TILES_DH = 8 * 2;    // Wdh  [128, 64]:  8 n-tiles x 2 k-blocks
static constexpr int TILES_TOTAL = TILES_IH + TILES_HH + TILES_DH;  // 304

#if defined(__HIP_DEVICE_COMPILE__) && \
    __has_builtin(__builtin_amdgcn_global_load_async_to_lds_b128) && \
    __has_builtin(__builtin_amdgcn_s_wait_asynccnt)
#define GRUD_ASYNC 1
#else
#define GRUD_ASYNC 0
#endif

#if GRUD_ASYNC
#define GRUD_AS1 __attribute__((address_space(1)))
#define GRUD_AS3 __attribute__((address_space(3)))
typedef int grud_v4i __attribute__((vector_size(16)));   // b128 payload type
__device__ inline void grud_async_b128(const float* g, float* l) {
  __builtin_amdgcn_global_load_async_to_lds_b128(
      (GRUD_AS1 grud_v4i*)g, (GRUD_AS3 grud_v4i*)l, /*offset=*/0, /*cpol=*/0);
}
// Each of the 256 threads issues one b128 per tensor: covers 16 rows x 64 f32.
__device__ inline void grud_issue_stage(
    const float* values, const float* masks, const float* deltas,
    const float* x_locf, float* buf, int b0, int t, int tid) {
  int r  = tid >> 4;           // 0..15
  int f4 = (tid & 15) << 2;    // 0,4,..,60
  size_t g = ((size_t)(b0 + r) * Tt + t) * Ff + f4;
  float* l = buf + r * Ff + f4;
  grud_async_b128(values + g, l);
  grud_async_b128(masks  + g, l + 16 * Ff);
  grud_async_b128(deltas + g, l + 2 * 16 * Ff);
  grud_async_b128(x_locf + g, l + 3 * 16 * Ff);
}
#endif

// ---------------------------------------------------------------------------
// Pack f32 weights into bf16 WMMA B-fragment layout:
//   B is 32x16 (KxN). Lane l holds column N = l&15; half e holds K = ((e&8)<<1) + ((l&16)>>1) + (e&7).
//   Fragment storage: packed[tile*512 + lane*16 + e]
// ---------------------------------------------------------------------------
__global__ __launch_bounds__(512) void grud_pack_kernel(
    const float* __restrict__ w_ih, const float* __restrict__ w_hh,
    const float* __restrict__ Wdh, __bf16* __restrict__ packed)
{
  int tile = blockIdx.x;
  int lane = threadIdx.x >> 4;   // 0..31
  int e    = threadIdx.x & 15;   // 0..15

  const float* W;
  int nt, kt, ldk;
  if (tile < TILES_IH)            { W = w_ih; nt = tile / 8; kt = tile % 8; ldk = INs; }
  else if (tile < TILES_IH + TILES_HH) {
    int tt = tile - TILES_IH;       W = w_hh; nt = tt / 4;   kt = tt % 4;   ldk = Hh;  }
  else {
    int tt = tile - TILES_IH - TILES_HH;
                                    W = Wdh;  nt = tt / 2;   kt = tt % 2;   ldk = Ff;  }

  int n = nt * 16 + (lane & 15);
  int k = kt * 32 + ((e & 8) << 1) + ((lane & 16) >> 1) + (e & 7);     // B-fragment K map
  packed[(size_t)tile * 512 + lane * 16 + e] = (__bf16)W[(size_t)n * ldk + k];
}

// ---------------------------------------------------------------------------
// Load a 16x32 bf16 A-fragment from LDS (row-major [16][ld], column offset k0).
// ISA A layout: lane l -> row M = l&15; kbase = (l&16)?8:0;
//   half e -> K = ((e&8)<<1) + kbase + (e&7)
// ---------------------------------------------------------------------------
__device__ inline v16bf load_A_lds(const __bf16* base, int ld, int k0, int lane) {
  int row = lane & 15;
  int kb  = (lane & 16) ? 8 : 0;
  const __bf16* p = base + row * ld + k0 + kb;
  v16bf a;
#pragma unroll
  for (int e = 0; e < 16; ++e)
    a[e] = p[((e & 8) << 1) + (e & 7)];
  return a;
}

__device__ inline v16bf load_B_packed(const __bf16* packed, int tile, int lane) {
  return *(const v16bf*)(packed + (size_t)tile * 512 + lane * 16);
}

__device__ inline v8f wmma_bf16(v16bf a, v16bf b, v8f c) {
  return __builtin_amdgcn_wmma_f32_16x16x32_bf16(
      /*neg_a=*/false, a, /*neg_b=*/false, b,
      /*c_mod=*/(short)0, c, /*reuse_a=*/false, /*reuse_b=*/false);
}

__device__ inline v8f splat8(float v) {
  v8f c;
#pragma unroll
  for (int i = 0; i < 8; ++i) c[i] = v;
  return c;
}

// ---------------------------------------------------------------------------
// Main persistent GRU-D kernel.
// ---------------------------------------------------------------------------
__global__ __launch_bounds__(256) void grud_main_kernel(
    const float* __restrict__ values, const float* __restrict__ masks,
    const float* __restrict__ deltas, const float* __restrict__ x_locf,
    const float* __restrict__ emp_mean,
    const float* __restrict__ Wdx, const float* __restrict__ bdx,
    const float* __restrict__ b_ih, const float* __restrict__ b_hh,
    const float* __restrict__ bdh,
    const float* __restrict__ Wc, const float* __restrict__ bc,
    const __bf16* __restrict__ packed, float* __restrict__ out)
{
  // LDS: activation matrices rebuilt every timestep.
  __shared__ __bf16 inp[16 * INs];   // [row][col]: x_rep(0:64) | h_dec(64:192) | m(192:256)
  __shared__ __bf16 dls[16 * Ff];    // d (deltas), bf16 A-source for gamma_h GEMM
  __shared__ float  hf[16 * Hh];     // f32 copy of h (for decay + final head)
  __shared__ float  wdxv[Ff], bdxv[Ff], empv[Ff];
#if GRUD_ASYNC
  __shared__ float  stage[2][4 * 16 * Ff];  // double-buffered async staging (32 KB)
#endif

  const int tid  = threadIdx.x;
  const int lane = tid & 31;
  const int w    = tid >> 5;          // wave 0..7
  const int b0   = blockIdx.x * 16;   // batch tile base

  if (tid < Ff) {
    wdxv[tid] = Wdx[tid * Ff + tid];  // diagonal of Wdx (diag=True TemporalDecay)
    bdxv[tid] = bdx[tid];
    empv[tid] = emp_mean[tid];
  }
  for (int i = tid; i < 16 * Hh; i += 256) hf[i] = 0.0f;
  for (int i = tid; i < 16 * INs; i += 256) inp[i] = (__bf16)0.0f;
  __syncthreads();

  // Per-wave fixed output column (H index) and folded biases.
  const int col  = lane & 15;
  const int hcol = w * 16 + col;                 // 0..127
  const float brz = b_ih[hcol]        + b_hh[hcol];
  const float bzz = b_ih[Hh + hcol]   + b_hh[Hh + hcol];
  const float bin = b_ih[2 * Hh + hcol];
  const float bhn = b_hh[2 * Hh + hcol];
  const float bdh_c = bdh[hcol];
  const int rbase = (lane & 16) ? 8 : 0;         // D-matrix row base per lane half

  const __bf16* Pih = packed;
  const __bf16* Phh = packed + (size_t)TILES_IH * 512;
  const __bf16* Pdh = packed + (size_t)(TILES_IH + TILES_HH) * 512;

  v8f hd = splat8(0.0f);   // decayed h for this wave's columns (f32)

#if GRUD_ASYNC
  grud_issue_stage(values, masks, deltas, x_locf, &stage[0][0], b0, 0, tid);
#endif

  for (int t = 0; t < Tt; ++t) {
    // ---- Stage x_rep, m, d into LDS -------------------------------------
#if GRUD_ASYNC
    const int cur = t & 1;
    __builtin_amdgcn_s_wait_asynccnt(0);
    __syncthreads();                       // all waves' DMA landed in stage[cur]
    if (t + 1 < Tt)
      grud_issue_stage(values, masks, deltas, x_locf,
                       &stage[cur ^ 1][0], b0, t + 1, tid);
    const float* sb = stage[cur];
    for (int i = tid; i < 16 * Ff; i += 256) {
      int r = i >> 6, f = i & 63;
      float x  = sb[i];
      float m  = sb[16 * Ff + i];
      float d  = sb[2 * 16 * Ff + i];
      float xl = sb[3 * 16 * Ff + i];
      float gx   = __expf(-fmaxf(0.0f, d * wdxv[f] + bdxv[f]));
      float xh   = gx * xl + (1.0f - gx) * empv[f];
      float xrep = m * x + (1.0f - m) * xh;
      inp[r * INs + f]        = (__bf16)xrep;
      inp[r * INs + 192 + f]  = (__bf16)m;
      dls[r * Ff + f]         = (__bf16)d;
    }
#else
    for (int i = tid; i < 16 * Ff; i += 256) {
      int r = i >> 6, f = i & 63;
      size_t g = ((size_t)(b0 + r) * Tt + t) * Ff + f;
      float x  = values[g];
      float m  = masks[g];
      float d  = deltas[g];
      float xl = x_locf[g];
      float gx   = __expf(-fmaxf(0.0f, d * wdxv[f] + bdxv[f]));
      float xh   = gx * xl + (1.0f - gx) * empv[f];
      float xrep = m * x + (1.0f - m) * xh;
      inp[r * INs + f]        = (__bf16)xrep;
      inp[r * INs + 192 + f]  = (__bf16)m;
      dls[r * Ff + f]         = (__bf16)d;
    }
#endif
    __syncthreads();

    // ---- gamma_h pre-activation: d @ Wdh.T + bdh (wave w -> n-tile w) -----
    v8f acc_g = splat8(bdh_c);
#pragma unroll
    for (int kt = 0; kt < 2; ++kt) {
      v16bf a = load_A_lds(dls, Ff, kt * 32, lane);
      v16bf b = load_B_packed(Pdh, w * 2 + kt, lane);
      acc_g = wmma_bf16(a, b, acc_g);
    }

    // ---- h_dec = h * exp(-relu(gamma_pre)); publish bf16 h_dec to inp -----
#pragma unroll
    for (int i = 0; i < 8; ++i) {
      float g = __expf(-fmaxf(0.0f, acc_g[i]));
      int row = rbase + i;
      float hdec = hf[row * Hh + hcol] * g;
      hd[i] = hdec;
      hf[row * Hh + hcol] = hdec;
      inp[row * INs + 64 + hcol] = (__bf16)hdec;
    }
    __syncthreads();

    // ---- Fused gate GEMMs -------------------------------------------------
    // acc_r = [x_rep|h|m] @ w_ih_r.T + h @ w_hh_r.T + (b_ih_r + b_hh_r)
    v8f acc_r  = splat8(brz);
    v8f acc_z  = splat8(bzz);
    v8f acc_in = splat8(bin);
    v8f acc_hn = splat8(bhn);
#pragma unroll
    for (int kt = 0; kt < 8; ++kt) {           // gi: K = 256
      v16bf a  = load_A_lds(inp, INs, kt * 32, lane);
      acc_r  = wmma_bf16(a, load_B_packed(Pih, (w     ) * 8 + kt, lane), acc_r );
      acc_z  = wmma_bf16(a, load_B_packed(Pih, (w +  8) * 8 + kt, lane), acc_z );
      acc_in = wmma_bf16(a, load_B_packed(Pih, (w + 16) * 8 + kt, lane), acc_in);
    }
#pragma unroll
    for (int kt = 0; kt < 4; ++kt) {           // gh: K = 128 (h_dec region)
      v16bf a  = load_A_lds(inp + 64, INs, kt * 32, lane);
      acc_r  = wmma_bf16(a, load_B_packed(Phh, (w     ) * 4 + kt, lane), acc_r );
      acc_z  = wmma_bf16(a, load_B_packed(Phh, (w +  8) * 4 + kt, lane), acc_z );
      acc_hn = wmma_bf16(a, load_B_packed(Phh, (w + 16) * 4 + kt, lane), acc_hn);
    }
    __syncthreads();   // all GEMM reads of inp done before h is overwritten

    // ---- Elementwise GRU update ------------------------------------------
#pragma unroll
    for (int i = 0; i < 8; ++i) {
      float r = 1.0f / (1.0f + __expf(-acc_r[i]));
      float z = 1.0f / (1.0f + __expf(-acc_z[i]));
      float n = tanhf(acc_in[i] + r * acc_hn[i]);
      float hn = (1.0f - z) * n + z * hd[i];
      int row = rbase + i;
      hf[row * Hh + hcol] = hn;
      inp[row * INs + 64 + hcol] = (__bf16)hn;
    }
    __syncthreads();   // h published before next step's staging/GEMMs
  }

  // ---- Head: sigmoid(h @ Wc.T + bc), C = 1 -------------------------------
  if (tid < 16) {
    float acc = bc[0];
#pragma unroll 8
    for (int j = 0; j < Hh; ++j) acc += hf[tid * Hh + j] * Wc[j];
    out[b0 + tid] = 1.0f / (1.0f + __expf(-acc));
  }
}

// ---------------------------------------------------------------------------
// Harness entry point.
// Input order: values, masks, deltas, x_locf, emp_mean, Wdh, bdh, Wdx, bdx,
//              w_ih, w_hh, b_ih, b_hh, Wc, bc
// ---------------------------------------------------------------------------
extern "C" void kernel_launch(void* const* d_in, const int* in_sizes, int n_in,
                              void* d_out, int out_size, void* d_ws, size_t ws_size,
                              hipStream_t stream) {
  const float* values   = (const float*)d_in[0];
  const float* masks    = (const float*)d_in[1];
  const float* deltas   = (const float*)d_in[2];
  const float* x_locf   = (const float*)d_in[3];
  const float* emp_mean = (const float*)d_in[4];
  const float* Wdh      = (const float*)d_in[5];
  const float* bdh      = (const float*)d_in[6];
  const float* Wdx      = (const float*)d_in[7];
  const float* bdx      = (const float*)d_in[8];
  const float* w_ih     = (const float*)d_in[9];
  const float* w_hh     = (const float*)d_in[10];
  const float* b_ih     = (const float*)d_in[11];
  const float* b_hh     = (const float*)d_in[12];
  const float* Wc       = (const float*)d_in[13];
  const float* bc       = (const float*)d_in[14];

  __bf16* packed = (__bf16*)d_ws;   // 304 KB of packed bf16 weight fragments

  grud_pack_kernel<<<TILES_TOTAL, 512, 0, stream>>>(w_ih, w_hh, Wdh, packed);
  grud_main_kernel<<<Bsz / 16, 256, 0, stream>>>(
      values, masks, deltas, x_locf, emp_mean, Wdx, bdx,
      b_ih, b_hh, bdh, Wc, bc, packed, (float*)d_out);
}
